// OptimizedMultiHeadAttention_36412732736211
// MI455X (gfx1250) — matrix operand order repaired and checked by reading, not compiled, not run on hardware
//
#include <hip/hip_runtime.h>
#include <hip/hip_bf16.h>
#include <math.h>

typedef __attribute__((ext_vector_type(16))) _Float16 v16h;
typedef __attribute__((ext_vector_type(8)))  _Float16 v8h;
typedef __attribute__((ext_vector_type(8)))  float    v8f;

// Problem constants (match reference)
constexpr int BB  = 2;
constexpr int TT  = 2048;
constexpr int CC  = 1600;
constexpr int HH  = 25;
constexpr int DH  = 64;
constexpr int M1  = BB * TT;      // 4096 rows
constexpr int N1  = 3 * CC;       // 4800 qkv cols
constexpr int K1  = CC;           // 1600 inner

// -------- fragment helpers (per CDNA5 WMMA VGPR layouts, wave32) --------
// A 16x32 f16: lane m = lane&15 ; k-half base kb = (lane<16)?0:8
//   a[i] = A[m][kb + (i<8 ? i : i+8)]
// B 32x16 f16: lane n = lane&15 ; k base kbB = (lane<16)?0:8
//   b[i] = B[kbB + (i<8 ? i : i+8)][n]
// C/D 16x16 f32: lane col n = lane&15 ; rows g8..g8+7, g8 = (lane<16)?0:8

__device__ inline v16h load_a_frag(const _Float16* __restrict__ row, int kb) {
    v8h lo = *(const v8h*)(row + kb);
    v8h hi = *(const v8h*)(row + kb + 16);
    v16h a;
#pragma unroll
    for (int i = 0; i < 8; ++i) { a[i] = lo[i]; a[8 + i] = hi[i]; }
    return a;
}

__device__ inline v16h load_b_frag_contig(const _Float16* __restrict__ col) {
    v8h lo = *(const v8h*)(col);
    v8h hi = *(const v8h*)(col + 16);
    v16h b;
#pragma unroll
    for (int i = 0; i < 8; ++i) { b[i] = lo[i]; b[8 + i] = hi[i]; }
    return b;
}

__device__ inline v8f wmma16(v16h a, v16h b, v8f c) {
    return __builtin_amdgcn_wmma_f32_16x16x32_f16(false, a, false, b, (short)0, c,
                                                  false, false);
}

// -------- CDNA5 async global->LDS copy (ASYNCcnt-tracked, per-lane 16B) -----
__device__ inline void async_copy_b128(const void* g, void* l) {
    // VDST = 32-bit LDS byte address (low 32 bits of the generic shared ptr),
    // VADDR = 64-bit global address. Tracked with ASYNCcnt.
    asm volatile("global_load_async_to_lds_b128 %0, %1, off"
                 :: "v"((unsigned)(unsigned long long)l), "v"(g)
                 : "memory");
}
__device__ inline void wait_async(int n) {
    if (n == 0)
        asm volatile("s_wait_asynccnt 0" ::: "memory");
    else
        asm volatile("s_wait_asynccnt 16" ::: "memory");
}

// ---------------- conversion kernels ----------------
__global__ void cvt_f32_to_f16(const float* __restrict__ in,
                               _Float16* __restrict__ out, int n) {
    int i = blockIdx.x * blockDim.x + threadIdx.x;
    if (i < n) out[i] = (_Float16)in[i];
}

// in[Kd][Nd] fp32 -> out[Nd][Kd] f16
__global__ void transpose_cvt(const float* __restrict__ in,
                              _Float16* __restrict__ out, int Kd, int Nd) {
    int k = blockIdx.x * blockDim.x + threadIdx.x;
    int n = blockIdx.y;
    if (k < Kd) out[(size_t)n * Kd + k] = (_Float16)in[(size_t)k * Nd + n];
}

// ---------------- QKV projection GEMM + head scatter ----------------
// Each wave: 32x64 output tile (2 A fragments x 4 B fragments -> 8 WMMAs/k-step).
// The 4 waves of a block take consecutive M-tiles with the SAME N-tile so the
// B stream is deduplicated by the WGP$. Epilogue scatters into head-major
// Q[b][h][t][dh], K[b][h][t][dh] and TRANSPOSED V[b][h][dh][t]; Q scaled.
__global__ __launch_bounds__(128) void qkv_gemm(
    const _Float16* __restrict__ xh, const _Float16* __restrict__ wt,
    const float* __restrict__ bias, _Float16* __restrict__ qh,
    _Float16* __restrict__ kh, _Float16* __restrict__ vt) {
    const int lane = threadIdx.x & 31;
    const int wid  = threadIdx.x >> 5;
    const int tn = blockIdx.x;                 // N-tile (64 cols)
    const int tm = blockIdx.y * 4 + wid;       // M-tile (32 rows)

    const int m   = lane & 15;
    const int kb  = (lane < 16) ? 0 : 8;
    const int kbB = (lane < 16) ? 0 : 8;
    const int nn  = lane & 15;
    const int g8  = (lane < 16) ? 0 : 8;

    v8f acc[2][4] = {{v8f{}, v8f{}, v8f{}, v8f{}}, {v8f{}, v8f{}, v8f{}, v8f{}}};
    const _Float16* arow0 = xh + (size_t)(tm * 32 + m) * K1;
    const _Float16* arow1 = arow0 + (size_t)16 * K1;

    for (int k0 = 0; k0 < K1; k0 += 32) {
        v16h a0 = load_a_frag(arow0 + k0, kb);
        v16h a1 = load_a_frag(arow1 + k0, kb);
#pragma unroll
        for (int s = 0; s < 4; ++s) {
            v16h b = load_b_frag_contig(
                wt + (size_t)(tn * 64 + s * 16 + nn) * K1 + k0 + kbB);
            acc[0][s] = wmma16(a0, b, acc[0][s]);
            acc[1][s] = wmma16(a1, b, acc[1][s]);
        }
    }

#pragma unroll
    for (int s = 0; s < 4; ++s) {
        const int d  = tn * 64 + s * 16 + nn;
        const float bv = bias[d];
        const int which = d / CC;
        const int c = d % CC;
        const int h = c / DH, dh = c % DH;
        const float scl = (which == 0) ? 0.125f : 1.0f;  // 1/sqrt(64)
#pragma unroll
        for (int a = 0; a < 2; ++a)
#pragma unroll
            for (int r = 0; r < 8; ++r) {
                const int row = tm * 32 + a * 16 + g8 + r;
                const int b_ = row >> 11;
                const int t_ = row & (TT - 1);
                const _Float16 val = (_Float16)((acc[a][s][r] + bv) * scl);
                if (which == 0)
                    qh[((size_t)(b_ * HH + h) * TT + t_) * DH + dh] = val;
                else if (which == 1)
                    kh[((size_t)(b_ * HH + h) * TT + t_) * DH + dh] = val;
                else  // V stored transposed: [b][h][dh][t]
                    vt[((size_t)(b_ * HH + h) * DH + dh) * TT + t_] = val;
            }
    }
}

// ---------------- flash attention (causal) ----------------
// One wave per (b,h, 16-query tile). 32-key blocks double-buffered into LDS
// via global_load_async_to_lds_b128 (ASYNCcnt), fragments read as ds_load_b128.
// Online softmax over 16-lane column groups; P staged through LDS into
// A-fragment layout; P*V via WMMA against transposed-V fragments.
__device__ inline void prefetch_kv(const _Float16* __restrict__ Kp,
                                   const _Float16* __restrict__ Vtp, int key0,
                                   _Float16* kdst, _Float16* vdst, int lane) {
    // K block: 32 contiguous rows of 128B -> 4KB contiguous
    const char* gk = (const char*)(Kp + (size_t)key0 * DH);
    char* lk = (char*)kdst;
#pragma unroll
    for (int it = 0; it < 8; ++it) {
        const int off = (it * 32 + lane) * 16;
        async_copy_b128(gk + off, lk + off);
    }
    // Vt block: 64 rows (dh) x 64B, global row stride TT halfs
    char* lv = (char*)vdst;
#pragma unroll
    for (int it = 0; it < 8; ++it) {
        const int idx = it * 32 + lane;          // 0..255
        const int row = idx >> 2, ch = (idx & 3) * 16;
        async_copy_b128((const char*)(Vtp + (size_t)row * TT + key0) + ch,
                        lv + row * 64 + ch);
    }
}

__global__ __launch_bounds__(128) void attn_fwd(
    const _Float16* __restrict__ qh, const _Float16* __restrict__ kh,
    const _Float16* __restrict__ vt, _Float16* __restrict__ yh) {
    __shared__ __align__(16) _Float16 skbuf[4][2][32 * DH];  // [wave][stage][key][dh]
    __shared__ __align__(16) _Float16 svbuf[4][2][DH * 32];  // [wave][stage][dh][key]
    __shared__ __align__(16) _Float16 spbuf[4][16 * 32];     // [wave][row][key]

    const int lane = threadIdx.x & 31;
    const int wid  = threadIdx.x >> 5;
    const int QT   = TT / 16;
    const int tile = blockIdx.x * 4 + wid;
    const int bh = tile / QT;
    const int qt = tile % QT;

    const _Float16* Qp  = qh + (size_t)bh * TT * DH;
    const _Float16* Kp  = kh + (size_t)bh * TT * DH;
    const _Float16* Vtp = vt + (size_t)bh * TT * DH;  // [dh][t]

    const int m   = lane & 15;
    const int kb  = (lane < 16) ? 0 : 8;
    const int kbB = (lane < 16) ? 0 : 8;
    const int nn  = lane & 15;
    const int g8  = (lane < 16) ? 0 : 8;

    v16h qa[2];
#pragma unroll
    for (int c = 0; c < 2; ++c)
        qa[c] = load_a_frag(Qp + (size_t)(qt * 16 + m) * DH + c * 32, kb);

    v8f yacc[4] = {v8f{}, v8f{}, v8f{}, v8f{}};
    float mrow[8], lrow[8];
#pragma unroll
    for (int r = 0; r < 8; ++r) { mrow[r] = -INFINITY; lrow[r] = 0.0f; }

    const int nblk = (qt * 16 + 16 + 31) / 32;  // causal reach in 32-key blocks
    prefetch_kv(Kp, Vtp, 0, skbuf[wid][0], svbuf[wid][0], lane);

    for (int j = 0; j < nblk; ++j) {
        const int key0 = j * 32;
        const int st = j & 1;
        if (j + 1 < nblk) {
            prefetch_kv(Kp, Vtp, key0 + 32, skbuf[wid][st ^ 1],
                        svbuf[wid][st ^ 1], lane);
            wait_async(16);  // block j's 16 async copies have landed in LDS
        } else {
            wait_async(0);
        }
        const _Float16* kbp = skbuf[wid][st];
        const _Float16* vbp = svbuf[wid][st];

        v8f s0 = {}, s1 = {};
#pragma unroll
        for (int c = 0; c < 2; ++c) {  // Dh chunks
            s0 = wmma16(qa[c], load_b_frag_contig(kbp + nn * DH + c * 32 + kbB), s0);
            s1 = wmma16(qa[c],
                        load_b_frag_contig(kbp + (16 + nn) * DH + c * 32 + kbB), s1);
        }

        float alpha[8];
#pragma unroll
        for (int r = 0; r < 8; ++r) {
            const int q = qt * 16 + g8 + r;
            float v0 = (key0 + nn      <= q) ? s0[r] : -INFINITY;
            float v1 = (key0 + 16 + nn <= q) ? s1[r] : -INFINITY;
            float v = fmaxf(v0, v1);
#pragma unroll
            for (int off = 1; off < 16; off <<= 1)
                v = fmaxf(v, __shfl_xor(v, off, 32));
            const float mnew = fmaxf(mrow[r], v);
            alpha[r] = __expf(mrow[r] - mnew);
            const float e0 = __expf(v0 - mnew);
            const float e1 = __expf(v1 - mnew);
            float bs = e0 + e1;
#pragma unroll
            for (int off = 1; off < 16; off <<= 1)
                bs += __shfl_xor(bs, off, 32);
            lrow[r] = lrow[r] * alpha[r] + bs;
            mrow[r] = mnew;
            spbuf[wid][(g8 + r) * 32 + nn]      = (_Float16)e0;
            spbuf[wid][(g8 + r) * 32 + 16 + nn] = (_Float16)e1;
        }
#pragma unroll
        for (int s = 0; s < 4; ++s)
#pragma unroll
            for (int r = 0; r < 8; ++r) yacc[s][r] *= alpha[r];

        // same-wave LDS RAW on the P tile (DS ops are in-order per wave)
        asm volatile("s_wait_dscnt 0" ::: "memory");

        v16h pa;
        {
            const _Float16* p = &spbuf[wid][m * 32 + kb];
#pragma unroll
            for (int i = 0; i < 8; ++i) { pa[i] = p[i]; pa[8 + i] = p[16 + i]; }
        }

        // Y += P * V  (Vt fragments: two 8-half runs per lane from LDS, at kbB and kbB+16)
#pragma unroll
        for (int s = 0; s < 4; ++s)
            yacc[s] = wmma16(pa, load_b_frag_contig(vbp + (s * 16 + nn) * 32 + kbB),
                             yacc[s]);
    }

    // epilogue: normalize and store head-major into yh [B][T][C]
    const int b_ = bh / HH, h = bh % HH;
#pragma unroll
    for (int s = 0; s < 4; ++s)
#pragma unroll
        for (int r = 0; r < 8; ++r) {
            const int t_ = qt * 16 + g8 + r;
            yh[(size_t)(b_ * TT + t_) * CC + h * DH + s * 16 + nn] =
                (_Float16)(yacc[s][r] / lrow[r]);
        }
}

// ---------------- output projection GEMM (writes fp32 d_out) ----------------
__global__ __launch_bounds__(128) void out_gemm(
    const _Float16* __restrict__ yh, const _Float16* __restrict__ wt,
    const float* __restrict__ bias, float* __restrict__ out) {
    const int lane = threadIdx.x & 31;
    const int wid  = threadIdx.x >> 5;
    const int tn = blockIdx.x;                 // 0..24
    const int tm = blockIdx.y * 4 + wid;       // 0..127

    const int m   = lane & 15;
    const int kb  = (lane < 16) ? 0 : 8;
    const int kbB = (lane < 16) ? 0 : 8;
    const int nn  = lane & 15;
    const int g8  = (lane < 16) ? 0 : 8;

    v8f acc[2][4] = {{v8f{}, v8f{}, v8f{}, v8f{}}, {v8f{}, v8f{}, v8f{}, v8f{}}};
    const _Float16* arow0 = yh + (size_t)(tm * 32 + m) * CC;
    const _Float16* arow1 = arow0 + (size_t)16 * CC;

    for (int k0 = 0; k0 < CC; k0 += 32) {
        v16h a0 = load_a_frag(arow0 + k0, kb);
        v16h a1 = load_a_frag(arow1 + k0, kb);
#pragma unroll
        for (int s = 0; s < 4; ++s) {
            v16h b = load_b_frag_contig(
                wt + (size_t)(tn * 64 + s * 16 + nn) * CC + k0 + kbB);
            acc[0][s] = wmma16(a0, b, acc[0][s]);
            acc[1][s] = wmma16(a1, b, acc[1][s]);
        }
    }

#pragma unroll
    for (int s = 0; s < 4; ++s) {
        const int d = tn * 64 + s * 16 + nn;
        const float bv = bias[d];
#pragma unroll
        for (int a = 0; a < 2; ++a)
#pragma unroll
            for (int r = 0; r < 8; ++r) {
                const int row = tm * 32 + a * 16 + g8 + r;
                out[(size_t)row * CC + d] = acc[a][s][r] + bv;
            }
    }
}

// ---------------- host-side launch ----------------
extern "C" void kernel_launch(void* const* d_in, const int* in_sizes, int n_in,
                              void* d_out, int out_size, void* d_ws,
                              size_t ws_size, hipStream_t stream) {
    const float* x     = (const float*)d_in[0];
    // d_in[1] = mask (causal, recomputed analytically — unused)
    const float* w_qkv = (const float*)d_in[2];
    const float* b_qkv = (const float*)d_in[3];
    const float* w_out = (const float*)d_in[4];
    const float* b_out = (const float*)d_in[5];
    float* out = (float*)d_out;

    char* ws = (char*)d_ws;
    size_t off = 0;
    auto alloc = [&](size_t bytes) -> void* {
        void* p = ws + off;
        off = (off + bytes + 255) & ~(size_t)255;
        return p;
    };
    _Float16* xh    = (_Float16*)alloc((size_t)M1 * K1 * 2);
    _Float16* wqkvt = (_Float16*)alloc((size_t)N1 * K1 * 2);
    _Float16* woutt = (_Float16*)alloc((size_t)CC * CC * 2);
    _Float16* qh    = (_Float16*)alloc((size_t)BB * HH * TT * DH * 2);
    _Float16* kh    = (_Float16*)alloc((size_t)BB * HH * TT * DH * 2);
    _Float16* vt    = (_Float16*)alloc((size_t)BB * HH * TT * DH * 2);
    _Float16* yh    = xh;  // xh dead after qkv_gemm -> reuse for attention out

    {
        int n = M1 * K1;
        cvt_f32_to_f16<<<(n + 255) / 256, 256, 0, stream>>>(x, xh, n);
    }
    transpose_cvt<<<dim3((K1 + 255) / 256, N1), 256, 0, stream>>>(w_qkv, wqkvt,
                                                                  K1, N1);
    transpose_cvt<<<dim3((CC + 255) / 256, CC), 256, 0, stream>>>(w_out, woutt,
                                                                  CC, CC);
    // QKV projection: grid = (N-tiles, M-supertiles of 4x32 rows)
    qkv_gemm<<<dim3(N1 / 64, M1 / 128), 128, 0, stream>>>(xh, wqkvt, b_qkv, qh,
                                                          kh, vt);
    // causal flash attention
    attn_fwd<<<BB * HH * (TT / 16) / 4, 128, 0, stream>>>(qh, kh, vt, yh);
    // output projection
    out_gemm<<<dim3(CC / 64, M1 / 128), 128, 0, stream>>>(yh, woutt, b_out, out);
}
